// SWA_63359357551387
// MI455X (gfx1250) — matrix operand order repaired and checked by reading, not compiled, not run on hardware
//
#include <hip/hip_runtime.h>

// ---------------------------------------------------------------------------
// CDNA5 (gfx1250) SWA block: bf16 WMMA GEMMs + flash sliding-window attention
// B=1, T=2048, DIM=2048, H=32, KV=8 (G=4), D=64, WIN=128
// ---------------------------------------------------------------------------

#define T_SEQ 2048
#define DIM_  2048
#define NH    32
#define NKV   8
#define HD    64
#define QKVN  3072   // (H + 2*KV) * D

typedef __attribute__((ext_vector_type(16))) __bf16 v16bf;
typedef __attribute__((ext_vector_type(8)))  __bf16 v8bf;
typedef __attribute__((ext_vector_type(8)))  float  v8f;
typedef __attribute__((ext_vector_type(4)))  int    v4i;
typedef __attribute__((address_space(1)))    v4i    v4i_glob;  // global (printed __device__)
typedef __attribute__((address_space(3)))    v4i    v4i_lds;   // LDS

#define WMMA_BF16(a, b, c) \
  __builtin_amdgcn_wmma_f32_16x16x32_bf16(false, (a), false, (b), (short)0, (c), false, false)

// ---- gfx1250 async global->LDS DMA (ASYNCcnt), with safe fallback ---------
#if defined(__has_builtin)
#if __has_builtin(__builtin_amdgcn_global_load_async_to_lds_b128)
#define HAVE_ASYNC_LDS 1
#endif
#endif
#ifndef HAVE_ASYNC_LDS
#define HAVE_ASYNC_LDS 0
#endif

#if HAVE_ASYNC_LDS
#if __has_builtin(__builtin_amdgcn_s_wait_asynccnt)
#define WAIT_ASYNC(n) __builtin_amdgcn_s_wait_asynccnt(n)
#else
#define WAIT_ASYNC(n) asm volatile("s_wait_asynccnt %0" ::"i"(n) : "memory")
#endif
#else
#define WAIT_ASYNC(n) ((void)0)
#endif

static __device__ inline v16bf cat8(v8bf lo, v8bf hi) {
  return __builtin_shufflevector(lo, hi, 0,1,2,3,4,5,6,7,8,9,10,11,12,13,14,15);
}
static __device__ inline v8bf ld8g(const __bf16* p) { return *(const v8bf*)p; }

static __device__ inline v8f zero8() {
  v8f z;
#pragma unroll
  for (int i = 0; i < 8; ++i) z[i] = 0.f;
  return z;
}

// stage 16 bytes global -> LDS (async DMA if available)
static __device__ inline void stage16(const __bf16* g, __bf16* l) {
#if HAVE_ASYNC_LDS
  __builtin_amdgcn_global_load_async_to_lds_b128((v4i_glob*)g, (v4i_lds*)l, 0, 0);
#else
  *(v8bf*)l = *(const v8bf*)g;
#endif
}

// A-fragment (16x32 bf16, row-major source, K contiguous), CDNA5 layout:
// lanes 0-15 : row = l15, elems 0-7 = K 0..7,  elems 8-15 = K 16..23
// lanes 16-31: row = l15, elems 0-7 = K 8..15, elems 8-15 = K 24..31
static __device__ inline v16bf load_afrag(const __bf16* rowk, int half) {
  return cat8(ld8g(rowk + half * 8), ld8g(rowk + 16 + half * 8));
}
// B-fragment (32x16 bf16): lane = column for both halves; half selects K 0-15/16-31.
static __device__ inline v16bf load_bfrag(const __bf16* p) {
  return cat8(ld8g(p), ld8g(p + 8));
}

// ---------------------------------------------------------------------------
// K1: fp32 -> bf16 convert (n multiple of 4)
// ---------------------------------------------------------------------------
__global__ __launch_bounds__(256) void k_cvt_bf16(const float* __restrict__ in,
                                                  __bf16* __restrict__ out, int n) {
  int i = (blockIdx.x * 256 + threadIdx.x) * 4;
  if (i >= n) return;
  float4 v = *(const float4*)(in + i);
  out[i + 0] = (__bf16)v.x;
  out[i + 1] = (__bf16)v.y;
  out[i + 2] = (__bf16)v.z;
  out[i + 3] = (__bf16)v.w;
}

// ---------------------------------------------------------------------------
// K2/K5: C[M,N] = A[M,K] * B[N,K]^T + bias[N]  (A,B bf16 K-contig, C fp32).
// Block = 8 waves = 512x64 tile; wave = 64x64 (4 A-frags x 4 B-frags,
// 16 WMMA per 32-K step).  B chunk (64 cols x 64 K) is double-buffered in
// LDS via async global->LDS DMA shared by all 8 waves (8x B-traffic cut).
// LDS row stride = 40 halves (80B) -> 16-lane x 16B fragment reads cover all
// 64 banks exactly twice (conflict-free wave32 pattern).
// ---------------------------------------------------------------------------
__global__ __launch_bounds__(256) void k_gemm_bf16_nt(const __bf16* __restrict__ A,
                                                      const __bf16* __restrict__ B,
                                                      const float* __restrict__ bias,
                                                      float* __restrict__ C,
                                                      int M, int N, int K) {
  __shared__ __bf16 bsh[2][2][64 * 40];   // [buffer][K-half(32)][64 cols x 40]
  const int lane = threadIdx.x & 31;
  const int wv   = threadIdx.x >> 5;
  const int ntn  = N >> 6;                 // 64-wide N tiles
  const int m0   = (blockIdx.x / ntn) * 512 + wv * 64;
  const int n0   = (blockIdx.x % ntn) * 64;
  const int half = lane >> 4;
  const int l15  = lane & 15;

  // staging assignment: thread -> (column 0..63, 8-half segment 0..3)
  const int  sc   = threadIdx.x >> 2;
  const int  sseg = threadIdx.x & 3;
  const __bf16* gB  = B + (size_t)(n0 + sc) * K + sseg * 8;
  const int     lof = sc * 40 + sseg * 8;

  v8f acc[4][4];
#pragma unroll
  for (int i = 0; i < 4; ++i)
#pragma unroll
    for (int j = 0; j < 4; ++j) acc[i][j] = zero8();

  const __bf16* pa[4];
#pragma unroll
  for (int mi = 0; mi < 4; ++mi) pa[mi] = A + (size_t)(m0 + mi * 16 + l15) * K;

  // prologue: stage chunk 0 (two 32-K halves)
  stage16(gB, &bsh[0][0][lof]);
  stage16(gB + 32, &bsh[0][1][lof]);

  const int nch = K >> 6;                  // 64-K chunks
  for (int i = 0; i < nch; ++i) {
    const int kc = i << 6;
    const int nb = i & 1;
    if (i + 1 < nch) {                     // prefetch next chunk into other buffer
      stage16(gB + kc + 64, &bsh[nb ^ 1][0][lof]);
      stage16(gB + kc + 96, &bsh[nb ^ 1][1][lof]);
      WAIT_ASYNC(2);                       // in-order: current chunk's 2 retired
    } else {
      WAIT_ASYNC(0);
    }
    __syncthreads();                       // chunk i visible to all waves

#pragma unroll
    for (int s = 0; s < 2; ++s) {          // two 32-K steps per chunk
      v16bf a0 = load_afrag(pa[0] + kc + s * 32, half);
      v16bf a1 = load_afrag(pa[1] + kc + s * 32, half);
      v16bf a2 = load_afrag(pa[2] + kc + s * 32, half);
      v16bf a3 = load_afrag(pa[3] + kc + s * 32, half);
#pragma unroll
      for (int j = 0; j < 4; ++j) {
        const __bf16* pbj = &bsh[nb][s][(j * 16 + l15) * 40 + half * 8];
        v16bf b = cat8(ld8g(pbj), ld8g(pbj + 16));
        acc[0][j] = WMMA_BF16(a0, b, acc[0][j]);
        acc[1][j] = WMMA_BF16(a1, b, acc[1][j]);
        acc[2][j] = WMMA_BF16(a2, b, acc[2][j]);
        acc[3][j] = WMMA_BF16(a3, b, acc[3][j]);
      }
    }
    __syncthreads();                       // all waves done reading buf nb
  }

#pragma unroll
  for (int mi = 0; mi < 4; ++mi) {
    const int rowb = m0 + mi * 16 + half * 8;
#pragma unroll
    for (int j = 0; j < 4; ++j) {
      const int col = n0 + j * 16 + l15;
      const float bs = bias[col];
      float* cp = C + (size_t)rowb * N + col;
#pragma unroll
      for (int r = 0; r < 8; ++r) cp[(size_t)r * N] = acc[mi][j][r] + bs;
    }
  }
}

// ---------------------------------------------------------------------------
// K3: RoPE + scatter.  qkv fp32 [T,3072] -> qb[T,H,D] bf16, kb[T,KV,D] bf16,
// vt[KV,D,T] bf16 (V transposed so PV B-fragments are contiguous loads).
// ---------------------------------------------------------------------------
__global__ __launch_bounds__(256) void k_rope_scatter(const float* __restrict__ qkv,
                                                      const float* __restrict__ cosT,
                                                      const float* __restrict__ sinT,
                                                      __bf16* __restrict__ qb,
                                                      __bf16* __restrict__ kb,
                                                      __bf16* __restrict__ vt) {
  int idx = blockIdx.x * 256 + threadIdx.x;   // T * 48 * 32 total
  int d   = idx & 31;
  int tmp = idx >> 5;
  int hh  = tmp % 48;
  int t   = tmp / 48;
  if (t >= T_SEQ) return;
  const float* row = qkv + (size_t)t * QKVN;
  const float c = cosT[t * HD + d];           // duplicated table: cos[t,d]==cos[t,d+32]
  const float s = sinT[t * HD + d];
  if (hh < NH) {                              // Q heads
    const int h = hh;
    float u1 = row[h * HD + d], u2 = row[h * HD + d + 32];
    __bf16* o = qb + ((size_t)t * NH + h) * HD;
    o[d]      = (__bf16)(u1 * c - u2 * s);
    o[d + 32] = (__bf16)(u2 * c + u1 * s);
  } else if (hh < NH + NKV) {                 // K heads
    const int g = hh - NH;
    float u1 = row[NH * HD + g * HD + d], u2 = row[NH * HD + g * HD + d + 32];
    __bf16* o = kb + ((size_t)t * NKV + g) * HD;
    o[d]      = (__bf16)(u1 * c - u2 * s);
    o[d + 32] = (__bf16)(u2 * c + u1 * s);
  } else {                                    // V heads -> transposed [KV,D,T]
    const int g = hh - NH - NKV;
    float v1 = row[(NH + NKV) * HD + g * HD + d];
    float v2 = row[(NH + NKV) * HD + g * HD + d + 32];
    vt[((size_t)g * HD + d) * T_SEQ + t]      = (__bf16)v1;
    vt[((size_t)g * HD + d + 32) * T_SEQ + t] = (__bf16)v2;
  }
}

// ---------------------------------------------------------------------------
// K4: flash sliding-window attention with sink.  One wave per (16-query tile,
// head).  9 key blocks of 16 cover the 144-key window; online softmax m/l
// initialized from the sink logit (NaN-free masking, sink in denominator).
// Row stats reduced with __shfl_xor over the 16-lane half that owns the row
// in the C-fragment layout.  P re-laid-out C->A via zero-padded 16x32 LDS.
// ---------------------------------------------------------------------------
__global__ __launch_bounds__(256) void k_swa_attn(const __bf16* __restrict__ qb,
                                                  const __bf16* __restrict__ kb,
                                                  const __bf16* __restrict__ vt,
                                                  const float* __restrict__ sinks,
                                                  __bf16* __restrict__ ob) {
  __shared__ __bf16 plds[8][16 * 32];
  const int lane = threadIdx.x & 31;
  const int wv   = threadIdx.x >> 5;
  const int widx = blockIdx.x * 8 + wv;      // (T/16)*H = 4096 waves
  const int h    = widx & (NH - 1);
  const int t0   = (widx >> 5) << 4;
  const int kv   = h >> 2;                   // G = 4
  const int half = lane >> 4;
  const int l15  = lane & 15;

  // zero pad region (cols 16..31) once; only cols 0..15 are ever rewritten
  for (int i = lane; i < 16 * 16; i += 32)
    plds[wv][(i >> 4) * 32 + 16 + (i & 15)] = (__bf16)0.f;

  const __bf16* qrow = qb + ((size_t)(t0 + l15) * NH + h) * HD;
  const v16bf aq0 = cat8(ld8g(qrow + half * 8),      ld8g(qrow + 16 + half * 8));
  const v16bf aq1 = cat8(ld8g(qrow + 32 + half * 8), ld8g(qrow + 48 + half * 8));

  const float snk = sinks[h];
  float m8[8], l8[8];
  v8f o[4];
#pragma unroll
  for (int r = 0; r < 8; ++r) { m8[r] = snk; l8[r] = 1.f; }
#pragma unroll
  for (int f = 0; f < 4; ++f) o[f] = zero8();

  const int myrow = t0 + half * 8;           // + r gives this lane's C-frag rows

  for (int j = 0; j < 9; ++j) {
    const int s0 = t0 - 128 + j * 16;
    // ---- scores: S = Q(16x64) . K^T(64x16) ----
    int sc = s0 + l15; sc = sc < 0 ? 0 : sc;            // clamped loads fully masked
    const __bf16* krow = kb + ((size_t)sc * NKV + kv) * HD;
    v16bf bk0 = cat8(ld8g(krow + half * 8),      ld8g(krow + 16 + half * 8));
    v16bf bk1 = cat8(ld8g(krow + 32 + half * 8), ld8g(krow + 48 + half * 8));
    v8f S = zero8();
    S = WMMA_BF16(aq0, bk0, S);
    S = WMMA_BF16(aq1, bk1, S);

    const int scol = s0 + l15;
#pragma unroll
    for (int r = 0; r < 8; ++r) {
      const int row = myrow + r;
      float x = S[r] * 0.125f;                           // D^-0.5
      const bool valid = (scol >= 0) & (scol <= row) & (scol >= row - 127);
      x = valid ? x : -1e30f;
      float mx = x;
      mx = fmaxf(mx, __shfl_xor(mx, 1));
      mx = fmaxf(mx, __shfl_xor(mx, 2));
      mx = fmaxf(mx, __shfl_xor(mx, 4));
      mx = fmaxf(mx, __shfl_xor(mx, 8));
      const float nm    = fmaxf(m8[r], mx);
      const float scale = __expf(m8[r] - nm);
      const float pe    = __expf(x - nm);
      float sum = pe;
      sum += __shfl_xor(sum, 1);
      sum += __shfl_xor(sum, 2);
      sum += __shfl_xor(sum, 4);
      sum += __shfl_xor(sum, 8);
      l8[r] = l8[r] * scale + sum;
      m8[r] = nm;
#pragma unroll
      for (int f = 0; f < 4; ++f) o[f][r] *= scale;
      plds[wv][(half * 8 + r) * 32 + l15] = (__bf16)pe;  // C-layout -> LDS tile
    }
    __builtin_amdgcn_wave_barrier();                     // order same-wave LDS RAW

    const __bf16* pl = &plds[wv][l15 * 32];
    v16bf pa = cat8(ld8g(pl + half * 16), ld8g(pl + half * 16 + 8));

    int key0 = s0 + half * 16;
    key0 = key0 < 0 ? 0 : key0;
    key0 = key0 > (T_SEQ - 16) ? (T_SEQ - 16) : key0;    // half=1 rows hit zero P
#pragma unroll
    for (int f = 0; f < 4; ++f) {
      const __bf16* vrow = vt + ((size_t)kv * HD + f * 16 + l15) * T_SEQ + key0;
      v16bf bv = load_bfrag(vrow);
      o[f] = WMMA_BF16(pa, bv, o[f]);
    }
  }

  float inv[8];
#pragma unroll
  for (int r = 0; r < 8; ++r) inv[r] = 1.f / l8[r];
#pragma unroll
  for (int f = 0; f < 4; ++f) {
    __bf16* op = ob + (size_t)myrow * DIM_ + h * HD + f * 16 + l15;
#pragma unroll
    for (int r = 0; r < 8; ++r) op[(size_t)r * DIM_] = (__bf16)(o[f][r] * inv[r]);
  }
}

// ---------------------------------------------------------------------------
// Launch
// ---------------------------------------------------------------------------
extern "C" void kernel_launch(void* const* d_in, const int* in_sizes, int n_in,
                              void* d_out, int out_size, void* d_ws, size_t ws_size,
                              hipStream_t stream) {
  const float* x    = (const float*)d_in[0];
  const float* cosT = (const float*)d_in[1];
  const float* sinT = (const float*)d_in[2];
  const float* Wqkv = (const float*)d_in[3];
  const float* bqkv = (const float*)d_in[4];
  const float* Wo   = (const float*)d_in[5];
  const float* bo   = (const float*)d_in[6];
  const float* snk  = (const float*)d_in[7];
  float* out = (float*)d_out;

  char* ws = (char*)d_ws;
  __bf16* xb    = (__bf16*)(ws);                       //  8 MiB (reused as ob)
  __bf16* wqkvb = (__bf16*)(ws + (size_t)( 8u << 20)); // 12 MiB
  __bf16* wob   = (__bf16*)(ws + (size_t)(20u << 20)); //  8 MiB
  float*  qkvf  = (float*) (ws + (size_t)(28u << 20)); // 24 MiB
  __bf16* qb    = (__bf16*)(ws + (size_t)(52u << 20)); //  8 MiB
  __bf16* kb    = (__bf16*)(ws + (size_t)(60u << 20)); //  2 MiB
  __bf16* vt    = (__bf16*)(ws + (size_t)(62u << 20)); //  2 MiB -> 64 MiB total
  __bf16* ob    = xb;                                  // xb dead after GEMM1

  // K1: fp32 -> bf16
  k_cvt_bf16<<<(T_SEQ * DIM_ / 4) / 256, 256, 0, stream>>>(x, xb, T_SEQ * DIM_);
  k_cvt_bf16<<<(QKVN * DIM_ / 4) / 256, 256, 0, stream>>>(Wqkv, wqkvb, QKVN * DIM_);
  k_cvt_bf16<<<(DIM_ * DIM_ / 4) / 256, 256, 0, stream>>>(Wo, wob, DIM_ * DIM_);

  // K2: qkv = x . Wqkv^T + bqkv   (512x64 block tiles: 4 x 48 blocks)
  k_gemm_bf16_nt<<<(T_SEQ / 512) * (QKVN / 64), 256, 0, stream>>>(
      xb, wqkvb, bqkv, qkvf, T_SEQ, QKVN, DIM_);

  // K3: RoPE + scatter/transpose
  k_rope_scatter<<<T_SEQ * 48 * 32 / 256, 256, 0, stream>>>(qkvf, cosT, sinT, qb, kb, vt);

  // K4: sliding-window flash attention with sink (4096 waves)
  k_swa_attn<<<(T_SEQ / 16) * NH / 8, 256, 0, stream>>>(qb, kb, vt, snk, ob);

  // K5: out = o . Wo^T + bo       (4 x 32 blocks)
  k_gemm_bf16_nt<<<(T_SEQ / 512) * (DIM_ / 64), 256, 0, stream>>>(
      ob, wob, bo, out, T_SEQ, DIM_, DIM_);
}
